// HiLo_75453985457224
// MI455X (gfx1250) — compile-verified, matrix operand order checked by reading, not run on hardware
//
#include <hip/hip_runtime.h>
#include <hip/hip_bf16.h>
#include <math.h>

// ---------------------------------------------------------------------------
// HiLo attention (LITv2-style) for MI455X / gfx1250, wave32 + WMMA bf16.
// All GEMMs and lofi attention use v_wmma_f32_16x16x32_bf16 (fp32 accum).
// GEMMs use 32x64 per-wave tiles: 12 b128 loads -> 8 WMMAs per k-step.
// ---------------------------------------------------------------------------

typedef __attribute__((ext_vector_type(16))) __bf16 v16bf;
typedef __attribute__((ext_vector_type(8)))  __bf16 v8bf;
typedef __attribute__((ext_vector_type(8)))  float  v8f;

#define ATTN_SCALE 0.17677669529663687f   // 32^-0.5

__device__ __forceinline__ v8f wmma_bf16(v16bf a, v16bf b, v8f c) {
  // D = A(16x32) * B^T-rows(16x32) + C, fp32 accumulate
  return __builtin_amdgcn_wmma_f32_16x16x32_bf16(false, a, false, b, (short)0, c,
                                                 false, false);
}

// Load one WMMA operand tile: caller passes pointer to this lane's row
// (row = tile_row_base + (lane&15)), k contiguous. Per the CDNA5 16-bit
// A-matrix layout: lanes 0-15 hold k = {0..7, 16..23}, lanes 16-31 hold
// k = {8..15, 24..31}  -> two 16-byte chunks at +koff and +16+koff.
__device__ __forceinline__ v16bf load_op(const __bf16* rowptr, int half) {
  const int koff = half ? 8 : 0;
  v8bf lo = *(const v8bf*)(rowptr + koff);
  v8bf hi = *(const v8bf*)(rowptr + 16 + koff);
  v16bf r;
#pragma unroll
  for (int i = 0; i < 8; ++i) { r[i] = lo[i]; r[8 + i] = hi[i]; }
  return r;
}

// Row reductions for the f32 C layout: row m lives on 16 lanes of one half
// (lane = n + 16*(m/8)); xor masks 1..8 never cross the 16-lane halves.
__device__ __forceinline__ float rowmax16(float v) {
#pragma unroll
  for (int m = 8; m >= 1; m >>= 1) v = fmaxf(v, __shfl_xor(v, m, 32));
  return v;
}
__device__ __forceinline__ float rowsum16(float v) {
#pragma unroll
  for (int m = 8; m >= 1; m >>= 1) v += __shfl_xor(v, m, 32);
  return v;
}

// ---------------------------------------------------------------------------
// Kernel 0: convert weights fp32 -> bf16, transposed to [n][k] row-major so
// every WMMA B operand is loaded k-contiguous exactly like the A operand.
// WtCat[512][256] = [h_qkv_w^T (384 rows) ; l_q_w^T (128 rows)]
// ---------------------------------------------------------------------------
__global__ void wconv_kernel(const float* __restrict__ h_qkv_w,
                             const float* __restrict__ l_q_w,
                             const float* __restrict__ l_kv_w,
                             const float* __restrict__ h_proj_w,
                             const float* __restrict__ l_proj_w,
                             __bf16* __restrict__ WtCat, __bf16* __restrict__ WtKV,
                             __bf16* __restrict__ WtHP, __bf16* __restrict__ WtLP) {
  int i = blockIdx.x * 256 + threadIdx.x;
  if (i < 512 * 256) {
    int n = i >> 8, k = i & 255;
    float v = (n < 384) ? h_qkv_w[k * 384 + n] : l_q_w[k * 128 + (n - 384)];
    WtCat[i] = (__bf16)v;
  } else if (i < 512 * 256 + 256 * 256) {
    int j = i - 512 * 256;
    int n = j >> 8, k = j & 255;
    WtKV[j] = (__bf16)l_kv_w[k * 256 + n];
  } else if (i < 512 * 256 + 256 * 256 + 128 * 128) {
    int j = i - (512 * 256 + 256 * 256);
    int n = j >> 7, k = j & 127;
    WtHP[j] = (__bf16)h_proj_w[k * 128 + n];
  } else if (i < 512 * 256 + 256 * 256 + 2 * 128 * 128) {
    int j = i - (512 * 256 + 256 * 256 + 128 * 128);
    int n = j >> 7, k = j & 127;
    WtLP[j] = (__bf16)l_proj_w[k * 128 + n];
  }
}

// ---------------------------------------------------------------------------
// Kernel 1: x [b,256,4096] fp32 -> Xbf [b,4096,256] bf16 (tiled transpose)
// ---------------------------------------------------------------------------
__global__ void pack_kernel(const float* __restrict__ x, __bf16* __restrict__ Xbf) {
  int bid = blockIdx.x;                 // 8 * 128 * 8 blocks
  int ct = bid & 7;                     // channel tile (32)
  int pt = (bid >> 3) & 127;            // pixel tile (32)
  int b  = bid >> 10;
  __shared__ float t[32][33];
  int tx = threadIdx.x, ty = threadIdx.y;        // (32, 8)
  const float* src = x + ((size_t)b * 256 + ct * 32) * 4096 + pt * 32;
#pragma unroll
  for (int i = 0; i < 4; ++i) t[ty + 8 * i][tx] = src[(size_t)(ty + 8 * i) * 4096 + tx];
  __syncthreads();
  __bf16* dst = Xbf + ((size_t)b * 4096 + pt * 32) * 256 + ct * 32;
#pragma unroll
  for (int i = 0; i < 4; ++i) dst[(size_t)(ty + 8 * i) * 256 + tx] = (__bf16)t[tx][ty + 8 * i];
}

// ---------------------------------------------------------------------------
// Kernel 2: 2x2 average pool  Xbf [b,4096,256] -> Pbf [b,1024,256]
// ---------------------------------------------------------------------------
__global__ void pool_kernel(const __bf16* __restrict__ Xbf, __bf16* __restrict__ Pbf) {
  int bm = blockIdx.x;                  // 8192 blocks, 256 threads (channel)
  int b = bm >> 10, m = bm & 1023;
  int my = m >> 5, mx = m & 31;
  int p = my * 128 + mx * 2;
  size_t base = ((size_t)b * 4096 + p) * 256 + threadIdx.x;
  float s = (float)Xbf[base] + (float)Xbf[base + 256] +
            (float)Xbf[base + 64 * 256] + (float)Xbf[base + 65 * 256];
  Pbf[(size_t)bm * 256 + threadIdx.x] = (__bf16)(0.25f * s);
}

// ---------------------------------------------------------------------------
// Kernel 3: fused QKV GEMM  Xbf[32768,256] x WtCat^T -> Hqkv[32768,384] bf16
//                                                    -> Lq  [32768,128] bf16
// One wave per 32x64 tile (2 A-tiles x 4 B-tiles, 8 accumulators):
// 12 b128 loads per k-step feed 8 WMMAs. 64 output cols per wave fall
// entirely in one destination (384 = 6*64). LDS-staged coalesced stores.
// ---------------------------------------------------------------------------
__global__ void gemm_qkv_kernel(const __bf16* __restrict__ X, const __bf16* __restrict__ Wt,
                                __bf16* __restrict__ Hqkv, __bf16* __restrict__ Lq) {
  const int lane = threadIdx.x & 31;
  const int wid  = threadIdx.x >> 5;
  const int wave = blockIdx.x * 8 + wid;       // 8192 waves
  const int nt = wave & 7;                     // 512 / 64
  const int mt = wave >> 3;                    // 32768 / 32
  const int half = lane >> 4, ln = lane & 15;
  const __bf16* arow0 = X + (size_t)(mt * 32 + ln) * 256;
  const __bf16* arow1 = arow0 + 16 * 256;
  const __bf16* brow  = Wt + (size_t)(nt * 64 + ln) * 256;
  v8f acc[8];
#pragma unroll
  for (int j = 0; j < 8; ++j) acc[j] = (v8f){0.f, 0.f, 0.f, 0.f, 0.f, 0.f, 0.f, 0.f};
#pragma unroll
  for (int kb = 0; kb < 256; kb += 32) {
    v16bf a0 = load_op(arow0 + kb, half);
    v16bf a1 = load_op(arow1 + kb, half);
#pragma unroll
    for (int j = 0; j < 4; ++j) {
      v16bf bj = load_op(brow + (size_t)j * 16 * 256 + kb, half);
      acc[2 * j]     = wmma_bf16(a0, bj, acc[2 * j]);
      acc[2 * j + 1] = wmma_bf16(a1, bj, acc[2 * j + 1]);
    }
  }
  __shared__ __bf16 st[8][16 * 16];
  __bf16* s = st[wid];
#pragma unroll
  for (int j = 0; j < 8; ++j) {        // tile j: cols (j>>1)*16, rows (j&1)*16
#pragma unroll
    for (int r = 0; r < 8; ++r) s[(r + 8 * half) * 16 + ln] = (__bf16)acc[j][r];
    int row  = mt * 32 + (j & 1) * 16 + (lane >> 1);
    int colo = (lane & 1) * 8;
    v8bf val = *(const v8bf*)(s + (lane >> 1) * 16 + colo);
    int c = nt * 64 + (j >> 1) * 16 + colo;
    if (nt < 6)
      *(v8bf*)(Hqkv + (size_t)row * 384 + c) = val;
    else
      *(v8bf*)(Lq + (size_t)row * 128 + (c - 384)) = val;
  }
}

// ---------------------------------------------------------------------------
// Kernel 4: KV GEMM  Pbf[8192,256] x WtKV^T -> Lk [b,1024,128] bf16 (rows)
//                                           -> LvT[b,4,32,1024] bf16 (V^T)
// 32x64 per-wave tiles; col groups: nt 0,1 -> K, nt 2,3 -> V (transposed).
// ---------------------------------------------------------------------------
__global__ void gemm_kv_kernel(const __bf16* __restrict__ P, const __bf16* __restrict__ Wt,
                               __bf16* __restrict__ Lk, __bf16* __restrict__ LvT) {
  const int lane = threadIdx.x & 31;
  const int wid  = threadIdx.x >> 5;
  const int wave = blockIdx.x * 8 + wid;       // 1024 waves
  const int nt = wave & 3;                     // 256 / 64
  const int mt = wave >> 2;                    // 8192 / 32
  const int half = lane >> 4, ln = lane & 15;
  const __bf16* arow0 = P + (size_t)(mt * 32 + ln) * 256;
  const __bf16* arow1 = arow0 + 16 * 256;
  const __bf16* brow  = Wt + (size_t)(nt * 64 + ln) * 256;
  v8f acc[8];
#pragma unroll
  for (int j = 0; j < 8; ++j) acc[j] = (v8f){0.f, 0.f, 0.f, 0.f, 0.f, 0.f, 0.f, 0.f};
#pragma unroll
  for (int kb = 0; kb < 256; kb += 32) {
    v16bf a0 = load_op(arow0 + kb, half);
    v16bf a1 = load_op(arow1 + kb, half);
#pragma unroll
    for (int j = 0; j < 4; ++j) {
      v16bf bj = load_op(brow + (size_t)j * 16 * 256 + kb, half);
      acc[2 * j]     = wmma_bf16(a0, bj, acc[2 * j]);
      acc[2 * j + 1] = wmma_bf16(a1, bj, acc[2 * j + 1]);
    }
  }
  __shared__ __bf16 st[8][16 * 16];
  if (nt < 2) {      // K half: row-major coalesced via LDS stage
    __bf16* s = st[wid];
#pragma unroll
    for (int j = 0; j < 8; ++j) {
#pragma unroll
      for (int r = 0; r < 8; ++r) s[(r + 8 * half) * 16 + ln] = (__bf16)acc[j][r];
      int row  = mt * 32 + (j & 1) * 16 + (lane >> 1);
      int colo = (lane & 1) * 8;
      v8bf val = *(const v8bf*)(s + (lane >> 1) * 16 + colo);
      *(v8bf*)(Lk + (size_t)row * 128 + nt * 64 + (j >> 1) * 16 + colo) = val;
    }
  } else {           // V half: write transposed (per-lane column scatter, 2 MB total)
#pragma unroll
    for (int j = 0; j < 8; ++j) {
      int c = nt * 64 + (j >> 1) * 16 + ln - 128;     // 0..127
      int head = c >> 5, d = c & 31;
      int gm = mt * 32 + (j & 1) * 16;
      int b = gm >> 10, m0 = gm & 1023;
      __bf16* dst = LvT + (size_t)((b * 4 + head) * 32 + d) * 1024 + m0;
#pragma unroll
      for (int r = 0; r < 8; ++r) dst[r + 8 * half] = (__bf16)acc[j][r];
    }
  }
}

// ---------------------------------------------------------------------------
// Kernel 5: hifi window attention. 4 tokens/window, 4x4 logits: far too small
// for WMMA -> one thread per (b, window, head), pure VALU fp32.
// ---------------------------------------------------------------------------
__global__ void hifi_attn_kernel(const __bf16* __restrict__ Hqkv, __bf16* __restrict__ HO) {
  int tid = blockIdx.x * blockDim.x + threadIdx.x;   // 32768
  int head = tid & 3;
  int g = (tid >> 2) & 1023;
  int b = tid >> 12;
  int gy = g >> 5, gx = g & 31;
  int p0 = gy * 128 + gx * 2;
  const __bf16* base = Hqkv + (size_t)b * 4096 * 384;
  int qc = head * 32, kc = 128 + head * 32, vc = 256 + head * 32;
  for (int i = 0; i < 4; ++i) {
    int pi = p0 + (i >> 1) * 64 + (i & 1);
    const __bf16* qp = base + (size_t)pi * 384 + qc;
    float q[32];
#pragma unroll
    for (int d = 0; d < 32; ++d) q[d] = (float)qp[d];
    float s[4];
#pragma unroll
    for (int j = 0; j < 4; ++j) {
      int pj = p0 + (j >> 1) * 64 + (j & 1);
      const __bf16* kp = base + (size_t)pj * 384 + kc;
      float acc = 0.f;
#pragma unroll
      for (int d = 0; d < 32; ++d) acc += q[d] * (float)kp[d];
      s[j] = acc * ATTN_SCALE;
    }
    float mx = fmaxf(fmaxf(s[0], s[1]), fmaxf(s[2], s[3]));
    float sum = 0.f;
#pragma unroll
    for (int j = 0; j < 4; ++j) { s[j] = __expf(s[j] - mx); sum += s[j]; }
    float inv = 1.f / sum;
    float o[32];
#pragma unroll
    for (int d = 0; d < 32; ++d) o[d] = 0.f;
#pragma unroll
    for (int j = 0; j < 4; ++j) {
      int pj = p0 + (j >> 1) * 64 + (j & 1);
      const __bf16* vp = base + (size_t)pj * 384 + vc;
      float pw = s[j] * inv;
#pragma unroll
      for (int d = 0; d < 32; ++d) o[d] += pw * (float)vp[d];
    }
    __bf16* op = HO + ((size_t)b * 4096 + pi) * 128 + qc;
#pragma unroll
    for (int d = 0; d < 32; ++d) op[d] = (__bf16)o[d];
  }
}

// ---------------------------------------------------------------------------
// Kernel 6: lofi attention, flash style. One wave per (b, head, 16-row Q tile).
// Per 32-key block: 2 WMMA (QK^T), online softmax (shfl row reductions),
// P repack via wave-private LDS (in-order DS pipe), 2 WMMA (P·V via V^T).
// Prefetch next key block (global_prefetch_b8).
// ---------------------------------------------------------------------------
__global__ void lofi_attn_kernel(const __bf16* __restrict__ Lq, const __bf16* __restrict__ Lk,
                                 const __bf16* __restrict__ LvT, __bf16* __restrict__ LO) {
  const int lane = threadIdx.x & 31;
  const int wid  = threadIdx.x >> 5;
  const int wave = blockIdx.x * 8 + wid;       // 8192 waves
  const int mt   = wave & 255;                 // 4096 / 16 q tiles
  const int head = (wave >> 8) & 3;
  const int b    = wave >> 10;
  const int half = lane >> 4, ln = lane & 15;

  const __bf16* qrow = Lq + (size_t)(b * 4096 + mt * 16 + ln) * 128 + head * 32;
  v16bf qa = load_op(qrow, half);

  v8f o_lo = {0.f, 0.f, 0.f, 0.f, 0.f, 0.f, 0.f, 0.f};
  v8f o_hi = {0.f, 0.f, 0.f, 0.f, 0.f, 0.f, 0.f, 0.f};
  float mrun[8], lrun[8];
#pragma unroll
  for (int r = 0; r < 8; ++r) { mrun[r] = -1e30f; lrun[r] = 0.f; }

  __shared__ __bf16 pl[8][16 * 32];
  __bf16* pbuf = pl[wid];

  const __bf16* kbase = Lk + (size_t)(b * 1024) * 128 + head * 32;
  const __bf16* vbase = LvT + (size_t)((b * 4 + head) * 32) * 1024;

  for (int kb = 0; kb < 1024; kb += 32) {
    v16bf k0 = load_op(kbase + (size_t)(kb + ln) * 128, half);
    v16bf k1 = load_op(kbase + (size_t)(kb + 16 + ln) * 128, half);
    if (kb + 32 < 1024) {
      __builtin_prefetch(kbase + (size_t)(kb + 32 + ln) * 128, 0, 3);
      __builtin_prefetch(kbase + (size_t)(kb + 48 + ln) * 128, 0, 3);
    }
    v8f z = {0.f, 0.f, 0.f, 0.f, 0.f, 0.f, 0.f, 0.f};
    v8f s0 = wmma_bf16(qa, k0, z);
    v8f s1 = wmma_bf16(qa, k1, z);

    float p0[8], p1[8], corr[8];
#pragma unroll
    for (int r = 0; r < 8; ++r) {
      float a  = s0[r] * ATTN_SCALE;
      float bq = s1[r] * ATTN_SCALE;
      float mx = rowmax16(fmaxf(a, bq));
      float nm = fmaxf(mrun[r], mx);
      float c  = __expf(mrun[r] - nm);
      float e0 = __expf(a - nm);
      float e1 = __expf(bq - nm);
      float rs = rowsum16(e0 + e1);
      lrun[r] = lrun[r] * c + rs;
      mrun[r] = nm;
      corr[r] = c;
      p0[r] = e0;
      p1[r] = e1;
    }
#pragma unroll
    for (int r = 0; r < 8; ++r) { o_lo[r] *= corr[r]; o_hi[r] *= corr[r]; }

    // repack P: C layout (row = r+8*half, col = ln / ln+16) -> A layout
#pragma unroll
    for (int r = 0; r < 8; ++r) {
      pbuf[(r + 8 * half) * 32 + ln]      = (__bf16)p0[r];
      pbuf[(r + 8 * half) * 32 + 16 + ln] = (__bf16)p1[r];
    }
    v16bf pa = load_op(pbuf + ln * 32, half);   // same-wave LDS: in-order DS pipe

    v16bf v0 = load_op(vbase + (size_t)ln * 1024 + kb, half);         // hd 0..15
    v16bf v1 = load_op(vbase + (size_t)(16 + ln) * 1024 + kb, half);  // hd 16..31
    o_lo = wmma_bf16(pa, v0, o_lo);
    o_hi = wmma_bf16(pa, v1, o_hi);
  }

#pragma unroll
  for (int r = 0; r < 8; ++r) {
    float inv = 1.f / lrun[r];
    o_lo[r] *= inv;
    o_hi[r] *= inv;
  }
  // stage O (16 rows x 32 cols bf16) in LDS, then coalesced 16B stores
#pragma unroll
  for (int r = 0; r < 8; ++r) {
    pbuf[(r + 8 * half) * 32 + ln]      = (__bf16)o_lo[r];
    pbuf[(r + 8 * half) * 32 + 16 + ln] = (__bf16)o_hi[r];
  }
  int row = mt * 16 + (lane >> 1);
  const __bf16* sp = pbuf + (lane >> 1) * 32 + (lane & 1) * 16;
  __bf16* dp = LO + ((size_t)b * 4096 + row) * 128 + head * 32 + (lane & 1) * 16;
  *(v8bf*)dp       = *(const v8bf*)sp;
  *(v8bf*)(dp + 8) = *(const v8bf*)(sp + 8);
}

// ---------------------------------------------------------------------------
// Kernel 7: dual projection + bias + concat + transpose to out [b,256,64,64].
// 32x64 per-wave tiles; col groups: nt 0,1 -> hifi proj, nt 2,3 -> lofi proj.
// Output written transposed (pixels contiguous) via LDS stage.
// ---------------------------------------------------------------------------
__global__ void proj_out_kernel(const __bf16* __restrict__ HO, const __bf16* __restrict__ LO,
                                const __bf16* __restrict__ WtHP, const __bf16* __restrict__ WtLP,
                                const float* __restrict__ hb, const float* __restrict__ lb,
                                float* __restrict__ out) {
  const int lane = threadIdx.x & 31;
  const int wid  = threadIdx.x >> 5;
  const int wave = blockIdx.x * 8 + wid;       // 4096 waves
  const int nt = wave & 3;                     // 256 / 64 channels
  const int mt = wave >> 2;                    // 32768 / 32 pixels
  const int half = lane >> 4, ln = lane & 15;

  const __bf16* src;
  const __bf16* wt;
  const float* bias;
  int wcol;
  if (nt < 2) { src = HO; wt = WtHP; bias = hb; wcol = nt * 64; }
  else        { src = LO; wt = WtLP; bias = lb; wcol = nt * 64 - 128; }

  const __bf16* arow0 = src + (size_t)(mt * 32 + ln) * 128;
  const __bf16* arow1 = arow0 + 16 * 128;
  const __bf16* brow  = wt + (size_t)(wcol + ln) * 128;
  v8f acc[8];
#pragma unroll
  for (int j = 0; j < 8; ++j) acc[j] = (v8f){0.f, 0.f, 0.f, 0.f, 0.f, 0.f, 0.f, 0.f};
#pragma unroll
  for (int kb = 0; kb < 128; kb += 32) {
    v16bf a0 = load_op(arow0 + kb, half);
    v16bf a1 = load_op(arow1 + kb, half);
#pragma unroll
    for (int j = 0; j < 4; ++j) {
      v16bf bj = load_op(brow + (size_t)j * 16 * 128 + kb, half);
      acc[2 * j]     = wmma_bf16(a0, bj, acc[2 * j]);
      acc[2 * j + 1] = wmma_bf16(a1, bj, acc[2 * j + 1]);
    }
  }
  __shared__ float st[8][16 * 17];
  float* s = st[wid];
#pragma unroll
  for (int j = 0; j < 8; ++j) {
    float bv = bias[wcol + (j >> 1) * 16 + ln];
    v8f a = acc[j];
#pragma unroll
    for (int r = 0; r < 8; ++r) a[r] += bv;
#pragma unroll
    for (int r = 0; r < 8; ++r) s[(r + 8 * half) * 17 + ln] = a[r];
    // transposed store: out[b][nt*64 + (j>>1)*16 + n][p0 + mh*8 .. +7]
    int gm = mt * 32 + (j & 1) * 16;
    int bb = gm >> 12;
    int p0 = gm & 4095;
    int och = nt * 64 + (j >> 1) * 16 + ln;
    float* op = out + ((size_t)bb * 256 + och) * 4096 + p0 + half * 8;
    float4 w0, w1;
    w0.x = s[(half * 8 + 0) * 17 + ln]; w0.y = s[(half * 8 + 1) * 17 + ln];
    w0.z = s[(half * 8 + 2) * 17 + ln]; w0.w = s[(half * 8 + 3) * 17 + ln];
    w1.x = s[(half * 8 + 4) * 17 + ln]; w1.y = s[(half * 8 + 5) * 17 + ln];
    w1.z = s[(half * 8 + 6) * 17 + ln]; w1.w = s[(half * 8 + 7) * 17 + ln];
    *(float4*)op       = w0;
    *(float4*)(op + 4) = w1;
  }
}

// ---------------------------------------------------------------------------
extern "C" void kernel_launch(void* const* d_in, const int* in_sizes, int n_in,
                              void* d_out, int out_size, void* d_ws, size_t ws_size,
                              hipStream_t stream) {
  (void)in_sizes; (void)n_in; (void)out_size; (void)ws_size;
  const float* x        = (const float*)d_in[0];
  const float* l_q_w    = (const float*)d_in[1];
  const float* l_kv_w   = (const float*)d_in[2];
  const float* l_proj_w = (const float*)d_in[3];
  const float* l_proj_b = (const float*)d_in[4];
  const float* h_qkv_w  = (const float*)d_in[5];
  const float* h_proj_w = (const float*)d_in[6];
  const float* h_proj_b = (const float*)d_in[7];
  float* out = (float*)d_out;

  char* ws = (char*)d_ws;
  size_t off = 0;
  auto alloc = [&](size_t bytes) {
    char* p = ws + off;
    off += (bytes + 255) & ~(size_t)255;
    return p;
  };
  __bf16* WtCat = (__bf16*)alloc(512 * 256 * 2);         // [h_qkv^T ; l_q^T]
  __bf16* WtKV  = (__bf16*)alloc(256 * 256 * 2);
  __bf16* WtHP  = (__bf16*)alloc(128 * 128 * 2);
  __bf16* WtLP  = (__bf16*)alloc(128 * 128 * 2);
  __bf16* Xbf   = (__bf16*)alloc((size_t)8 * 4096 * 256 * 2);
  __bf16* Pbf   = (__bf16*)alloc((size_t)8 * 1024 * 256 * 2);
  __bf16* Hqkv  = (__bf16*)alloc((size_t)8 * 4096 * 384 * 2);
  __bf16* Lqb   = (__bf16*)alloc((size_t)8 * 4096 * 128 * 2);
  __bf16* Lk    = (__bf16*)alloc((size_t)8 * 1024 * 128 * 2);
  __bf16* LvT   = (__bf16*)alloc((size_t)8 * 4 * 32 * 1024 * 2);
  __bf16* HO    = (__bf16*)alloc((size_t)8 * 4096 * 128 * 2);
  __bf16* LO    = (__bf16*)alloc((size_t)8 * 4096 * 128 * 2);

  wconv_kernel<<<896, 256, 0, stream>>>(h_qkv_w, l_q_w, l_kv_w, h_proj_w, l_proj_w,
                                        WtCat, WtKV, WtHP, WtLP);
  pack_kernel<<<8192, dim3(32, 8), 0, stream>>>(x, Xbf);
  pool_kernel<<<8192, 256, 0, stream>>>(Xbf, Pbf);
  gemm_qkv_kernel<<<1024, 256, 0, stream>>>(Xbf, WtCat, Hqkv, Lqb);
  gemm_kv_kernel<<<128, 256, 0, stream>>>(Pbf, WtKV, Lk, LvT);
  hifi_attn_kernel<<<128, 256, 0, stream>>>(Hqkv, HO);
  lofi_attn_kernel<<<1024, 256, 0, stream>>>(Lqb, Lk, LvT, LO);
  proj_out_kernel<<<512, 256, 0, stream>>>(HO, LO, WtHP, WtLP, h_proj_b, l_proj_b, out);
}
